// LlamaAttention_36223754174535
// MI455X (gfx1250) — matrix operand order repaired and checked by reading, not compiled, not run on hardware
//
#include <hip/hip_runtime.h>
#include <hip/hip_bf16.h>

// ---------------------------------------------------------------------------
// LlamaAttention forward for MI455X (gfx1250, wave32, WMMA).
//   B=2, S=2048, D=2048, H=32, HD=64
// Strategy: bf16 operands + f32 accumulate via v_wmma_f32_16x16x32_bf16.
// CDNA5 data movement: TENSOR_LOAD_TO_LDS (TDM) for attention K tiles,
// GLOBAL_LOAD_ASYNC_TO_LDS_B128 for the bf16 A tiles of the final GEMM,
// global_prefetch for next-K GEMM tiles.
// ---------------------------------------------------------------------------

#define Bsz 2
#define Ssz 2048
#define Dsz 2048
#define Hsz 32
#define HDsz 64
#define Msz (Bsz * Ssz)   // 4096 rows

typedef __attribute__((ext_vector_type(16))) __bf16 v16bf;
typedef __attribute__((ext_vector_type(8)))  float  v8f;
typedef __attribute__((ext_vector_type(4)))  unsigned int v4u;
typedef __attribute__((ext_vector_type(8)))  int v8i_t;
typedef __attribute__((ext_vector_type(4)))  int v4i_t;

union ABFrag { unsigned short u[16]; v16bf v; };

__device__ __forceinline__ unsigned short f2bf(float f) {
    unsigned int u = __float_as_uint(f);
    return (unsigned short)((u + 0x7FFFu + ((u >> 16) & 1u)) >> 16);  // RNE
}

// Low 32 bits of a generic pointer to LDS = wave-relative LDS byte offset
// (ISA 10.2: LDS aperture).
__device__ __forceinline__ unsigned int lds_off(const void* p) {
    return (unsigned int)(unsigned long long)(size_t)p;
}

// Per-lane async copy of 16 bytes global -> LDS (ASYNCcnt-tracked).
__device__ __forceinline__ void async_copy_b128(unsigned int dst_lds, const void* src) {
    asm volatile("global_load_async_to_lds_b128 %0, %1, off"
                 :: "v"(dst_lds), "v"((unsigned long long)src)
                 : "memory");
}
__device__ __forceinline__ void wait_async0() {
    asm volatile("s_wait_asynccnt 0x0" ::: "memory");
}

// ---------------------------------------------------------------------------
// GEMM: C[M,N] = A[M,K] * W[N,K]^T.  A is f32 (A_BF16=0) or bf16 (A_BF16=1).
// Block tile 128x128, K-step 32. 256 threads = 8 waves as 4(m) x 2(n), each
// wave computes 2x4 subtiles of 16x16 (wave tile 32x64).
// ---------------------------------------------------------------------------
template <int A_BF16>
__global__ __launch_bounds__(256) void gemm_xt(
    const void* __restrict__ Aptr,
    const float* __restrict__ Wptr,
    float* __restrict__ Cf,            // optional f32 output
    unsigned short* __restrict__ Cb,   // optional bf16 output
    int Mdim, int Ndim, int Kdim) {

    __shared__ unsigned int As[128 * 16];  // 128 rows x 32 bf16 (pairs)
    __shared__ unsigned int Bs[128 * 16];

    const int tid  = threadIdx.x;
    const int lane = tid & 31;
    const int wave = tid >> 5;
    const int wm   = wave & 3;   // 0..3 -> m offset wm*32
    const int wn   = wave >> 2;  // 0..1 -> n offset wn*64
    const int mr   = lane & 15;
    const int kg   = lane >> 4;  // lane half
    const int bm   = blockIdx.y * 128;
    const int bn   = blockIdx.x * 128;
    (void)Mdim;

    v8f acc[2][4] = {};

    for (int k0 = 0; k0 < Kdim; k0 += 32) {
        // ---- stage A tile (128x32) into LDS as bf16 ----
        if (A_BF16) {
            // pure bf16 copy -> CDNA5 async global->LDS path
            const unsigned short* Ab = (const unsigned short*)Aptr;
            int row = tid >> 1;
            int c   = (tid & 1) * 16;  // column 0 or 16; 8 bf16 per 16-byte transfer
            async_copy_b128(lds_off(&As[row * 16 + (c >> 1)]),
                            Ab + (size_t)(bm + row) * Kdim + k0 + c);
        } else {
            const float* Af = (const float*)Aptr;
#pragma unroll
            for (int j = 0; j < 4; ++j) {
                int l4  = tid + j * 256;          // 1024 float4 total
                int row = l4 >> 3;
                int c4  = (l4 & 7) * 4;
                float4 d = *(const float4*)(Af + (size_t)(bm + row) * Kdim + k0 + c4);
                As[row * 16 + (c4 >> 1)]     = ((unsigned)f2bf(d.y) << 16) | f2bf(d.x);
                As[row * 16 + (c4 >> 1) + 1] = ((unsigned)f2bf(d.w) << 16) | f2bf(d.z);
            }
        }
        // ---- stage W tile (128x32) into LDS as bf16 (needs f32->bf16 cvt) --
#pragma unroll
        for (int j = 0; j < 4; ++j) {
            int l4  = tid + j * 256;
            int row = l4 >> 3;
            int c4  = (l4 & 7) * 4;
            float4 d = *(const float4*)(Wptr + (size_t)(bn + row) * Kdim + k0 + c4);
            Bs[row * 16 + (c4 >> 1)]     = ((unsigned)f2bf(d.y) << 16) | f2bf(d.x);
            Bs[row * 16 + (c4 >> 1) + 1] = ((unsigned)f2bf(d.w) << 16) | f2bf(d.z);
        }
        if (A_BF16) wait_async0();   // drain ASYNCcnt before publishing tile
        __syncthreads();

        // gfx1250 prefetch of next K tile (global_prefetch_b8)
        if (k0 + 32 < Kdim) {
            __builtin_prefetch(Wptr + (size_t)(bn + (tid >> 1)) * Kdim + k0 + 32, 0, 1);
            if (!A_BF16)
                __builtin_prefetch((const float*)Aptr + (size_t)(bm + (tid >> 1)) * Kdim + k0 + 32, 0, 1);
        }

        // ---- fragments per ISA 7.12.2 layouts ----
        ABFrag afrag[2], bfrag[4];
#pragma unroll
        for (int i = 0; i < 2; ++i) {
            const unsigned short* rowp =
                (const unsigned short*)&As[(wm * 32 + i * 16 + mr) * 16];
#pragma unroll
            for (int e = 0; e < 8; ++e) afrag[i].u[e]     = rowp[kg * 8 + e];
#pragma unroll
            for (int e = 0; e < 8; ++e) afrag[i].u[8 + e] = rowp[16 + kg * 8 + e];
        }
#pragma unroll
        for (int i = 0; i < 4; ++i) {
            const unsigned short* rowp =
                (const unsigned short*)&Bs[(wn * 64 + i * 16 + mr) * 16];
#pragma unroll
            for (int e = 0; e < 16; ++e) bfrag[i].u[e] = rowp[(e >> 3) * 16 + kg * 8 + (e & 7)];
        }
#pragma unroll
        for (int i = 0; i < 2; ++i)
#pragma unroll
            for (int jn = 0; jn < 4; ++jn)
                acc[i][jn] = __builtin_amdgcn_wmma_f32_16x16x32_bf16(
                    false, afrag[i].v, false, bfrag[jn].v,
                    (short)0, acc[i][jn], false, false);
        __syncthreads();
    }

    // ---- store: C/D layout -> lane holds n=lane&15, vgpr j -> m = j + half*8
    const int mh = (lane >> 4) * 8;
#pragma unroll
    for (int i = 0; i < 2; ++i)
#pragma unroll
        for (int jn = 0; jn < 4; ++jn) {
            int n = bn + wn * 64 + jn * 16 + mr;
#pragma unroll
            for (int r = 0; r < 8; ++r) {
                int m = bm + wm * 32 + i * 16 + mh + r;
                float val = acc[i][jn][r];
                if (Cf) Cf[(size_t)m * Ndim + n] = val;
                if (Cb) Cb[(size_t)m * Ndim + n] = f2bf(val);
            }
        }
}

// ---------------------------------------------------------------------------
// Flash-style attention. One block = 1 (b,h) x 64 query rows; 4 waves x 16
// rows. Iterate over 64-key tiles with online softmax; QK^T and PV in WMMA.
// K tiles arrive via the Tensor Data Mover (2D tile descriptor, TENSORcnt).
// Q/K/V/O are bf16 in [B,S,H,HD] layout.
// ---------------------------------------------------------------------------
__global__ __launch_bounds__(128) void attn_fwd(
    const unsigned short* __restrict__ Qb,
    const unsigned short* __restrict__ Kb,
    const unsigned short* __restrict__ Vb,
    unsigned short* __restrict__ Ob) {

    __shared__ unsigned int Ks[64 * 32];        // [key][hd] bf16
    __shared__ unsigned int VsT[64 * 32];       // [hd][key] bf16 (transposed)
    __shared__ unsigned short Ps[4][16][64];    // per-wave P tile [m][key]

    const int tid  = threadIdx.x;
    const int lane = tid & 31;
    const int wave = tid >> 5;
    const int mr   = lane & 15;
    const int kg   = lane >> 4;
    const int bh   = blockIdx.y;
    const int b    = bh >> 5;
    const int h    = bh & 31;
    const int qbase = blockIdx.x * 64;

    // Q A-fragments held in registers for the whole kernel (HD=64 -> 2 k-steps)
    ABFrag qf[2];
    {
        const int qrow = qbase + wave * 16 + mr;
        const unsigned short* qrp = Qb + ((size_t)(b * Ssz + qrow) * Hsz + h) * HDsz;
#pragma unroll
        for (int ks = 0; ks < 2; ++ks) {
#pragma unroll
            for (int e = 0; e < 8; ++e) qf[ks].u[e]     = qrp[ks * 32 + kg * 8 + e];
#pragma unroll
            for (int e = 0; e < 8; ++e) qf[ks].u[8 + e] = qrp[ks * 32 + 16 + kg * 8 + e];
        }
    }

    float rmax[8], rsum[8];
#pragma unroll
    for (int r = 0; r < 8; ++r) { rmax[r] = -3.0e38f; rsum[r] = 0.0f; }
    v8f o[4] = {};

    for (int kt = 0; kt < Ssz / 64; ++kt) {
        const int keybase = kt * 64;
        const unsigned short* ktile =
            Kb + ((size_t)(b * Ssz + keybase) * Hsz + h) * HDsz;

#if __has_builtin(__builtin_amdgcn_tensor_load_to_lds)
        // ---- TDM: one wave issues a 2D tile descriptor (64x64 bf16,
        //      row stride D elements) Global -> LDS; TENSORcnt-tracked.
        if (wave == 0) {
            unsigned long long ga = (unsigned long long)(const void*)ktile;
            v4u g0 = {1u,                                  // count=1, user mode
                      lds_off(Ks),                          // lds_addr [63:32]
                      (unsigned int)ga,                     // global_addr lo
                      ((unsigned int)(ga >> 32) & 0x01FFFFFFu) | 0x80000000u}; // hi + type=2
            v8i_t g1 = {(int)(1u << 16),   // data_size=1 (2 bytes/elem)
                        (int)(64u << 16),  // tensor_dim0 = 64      (bits 79:48)
                        (int)(64u << 16),  // tensor_dim1 = 64      (bits 111:80)
                        (int)(64u << 16),  // tile_dim0   = 64      (bits 127:112)
                        64,                // tile_dim1   = 64      (bits 143:128)
                        2048,              // tensor_dim0_stride=D  (bits 207:160)
                        0, 0};
            v4i_t gz = {0, 0, 0, 0};
            v8i_t gz8 = {0, 0, 0, 0, 0, 0, 0, 0};
            // 6-arg variant (this toolchain): (g0, g1, g2, g3, g4, cpol)
            __builtin_amdgcn_tensor_load_to_lds(g0, g1, gz, gz, gz8, 0);
        }
#else
        // fallback: per-lane vector copy of K tile
#pragma unroll
        for (int j = 0; j < 4; ++j) {
            int l   = tid + j * 128;
            int row = l >> 3;
            int c8  = l & 7;
            uint4 d = *(const uint4*)(ktile + (size_t)row * Hsz * HDsz + c8 * 8);
            *(uint4*)&Ks[row * 32 + c8 * 4] = d;
        }
#endif

        // ---- stage V tile transposed: VsT[hd][key] (overlaps TDM) ----
#pragma unroll
        for (int j = 0; j < 4; ++j) {
            int l   = tid + j * 128;
            int row = l >> 3;            // key
            int c8  = l & 7;             // hd block
            uint4 d = *(const uint4*)(Vb + ((size_t)(b * Ssz + keybase + row) * Hsz + h) * HDsz + c8 * 8);
            unsigned short* vt = (unsigned short*)VsT;
            unsigned int w[4] = {d.x, d.y, d.z, d.w};
            int hdb = c8 * 8;
#pragma unroll
            for (int e = 0; e < 8; ++e)
                vt[(hdb + e) * 64 + row] =
                    (unsigned short)((w[e >> 1] >> ((e & 1) * 16)) & 0xFFFFu);
        }
#if __has_builtin(__builtin_amdgcn_tensor_load_to_lds)
        if (wave == 0) __builtin_amdgcn_s_wait_tensorcnt(0);
#endif
        __syncthreads();

        // ---- S = (Q K^T) / sqrt(HD), 4 n-tiles of 16 keys ----
        v8f sf[4];
#pragma unroll
        for (int nt = 0; nt < 4; ++nt) {
            v8f s = {};
            ABFrag bf;
#pragma unroll
            for (int ks = 0; ks < 2; ++ks) {
                const unsigned short* rowp = (const unsigned short*)&Ks[(nt * 16 + mr) * 32];
#pragma unroll
                for (int e = 0; e < 16; ++e) bf.u[e] = rowp[ks * 32 + (e >> 3) * 16 + kg * 8 + (e & 7)];
                s = __builtin_amdgcn_wmma_f32_16x16x32_bf16(
                    false, qf[ks].v, false, bf.v, (short)0, s, false, false);
            }
#pragma unroll
            for (int r = 0; r < 8; ++r) s[r] *= 0.125f;  // 1/sqrt(64)
            sf[nt] = s;
        }

        // ---- online softmax: row max across 4 tiles + 16 lanes of each half
        float tmax[8];
#pragma unroll
        for (int r = 0; r < 8; ++r)
            tmax[r] = fmaxf(fmaxf(sf[0][r], sf[1][r]), fmaxf(sf[2][r], sf[3][r]));
#pragma unroll
        for (int off = 1; off <= 8; off <<= 1)
#pragma unroll
            for (int r = 0; r < 8; ++r)
                tmax[r] = fmaxf(tmax[r], __shfl_xor(tmax[r], off, 32));

#pragma unroll
        for (int r = 0; r < 8; ++r) {
            float nm   = fmaxf(rmax[r], tmax[r]);
            float corr = __expf(rmax[r] - nm);
            rmax[r] = nm;
            rsum[r] *= corr;
#pragma unroll
            for (int nt = 0; nt < 4; ++nt) o[nt][r] *= corr;
        }

        // ---- P = exp(S - max); partial row sums; spill P to LDS (A-layout)
        float psum[8] = {0, 0, 0, 0, 0, 0, 0, 0};
#pragma unroll
        for (int nt = 0; nt < 4; ++nt)
#pragma unroll
            for (int r = 0; r < 8; ++r) {
                float p = __expf(sf[nt][r] - rmax[r]);
                psum[r] += p;
                Ps[wave][r + kg * 8][nt * 16 + mr] = f2bf(p);
            }
#pragma unroll
        for (int off = 1; off <= 8; off <<= 1)
#pragma unroll
            for (int r = 0; r < 8; ++r) psum[r] += __shfl_xor(psum[r], off, 32);
#pragma unroll
        for (int r = 0; r < 8; ++r) rsum[r] += psum[r];

        // ---- O += P @ V ----
#pragma unroll
        for (int ks = 0; ks < 2; ++ks) {
            ABFrag pf;
            const unsigned short* prow = &Ps[wave][mr][0];
#pragma unroll
            for (int e = 0; e < 8; ++e) pf.u[e]     = prow[ks * 32 + kg * 8 + e];
#pragma unroll
            for (int e = 0; e < 8; ++e) pf.u[8 + e] = prow[ks * 32 + 16 + kg * 8 + e];
#pragma unroll
            for (int nt = 0; nt < 4; ++nt) {
                ABFrag vf;
                const unsigned short* vrow = (const unsigned short*)&VsT[(nt * 16 + mr) * 32];
#pragma unroll
                for (int e = 0; e < 16; ++e) vf.u[e] = vrow[ks * 32 + (e >> 3) * 16 + kg * 8 + (e & 7)];
                o[nt] = __builtin_amdgcn_wmma_f32_16x16x32_bf16(
                    false, pf.v, false, vf.v, (short)0, o[nt], false, false);
            }
        }
        __syncthreads();
    }

    // ---- finalize: O /= rowsum, write bf16 [B,S,H,HD] ----
#pragma unroll
    for (int r = 0; r < 8; ++r) {
        float inv = 1.0f / rsum[r];
        int q = qbase + wave * 16 + r + kg * 8;
        unsigned short* orow = Ob + ((size_t)(b * Ssz + q) * Hsz + h) * HDsz;
#pragma unroll
        for (int nt = 0; nt < 4; ++nt)
            orow[nt * 16 + mr] = f2bf(o[nt][r] * inv);
    }
}

// ---------------------------------------------------------------------------
// Host launcher. d_out = [out (B*S*D) | k (B*S*D) | v (B*S*D)] f32.
// Workspace: Qb,Kb,Vb,Ab bf16 each B*S*D -> 4*8388608*2 = 64 MiB.
// ---------------------------------------------------------------------------
extern "C" void kernel_launch(void* const* d_in, const int* in_sizes, int n_in,
                              void* d_out, int out_size, void* d_ws, size_t ws_size,
                              hipStream_t stream) {
    (void)in_sizes; (void)n_in; (void)out_size; (void)ws_size;
    const float* X  = (const float*)d_in[0];
    const float* Wq = (const float*)d_in[1];
    const float* Wk = (const float*)d_in[2];
    const float* Wv = (const float*)d_in[3];
    const float* Wo = (const float*)d_in[4];

    const size_t BSD = (size_t)Bsz * Ssz * Dsz;
    float* out  = (float*)d_out;
    float* kout = out + BSD;
    float* vout = kout + BSD;

    unsigned short* Qb = (unsigned short*)d_ws;
    unsigned short* Kb = Qb + BSD;
    unsigned short* Vb = Kb + BSD;
    unsigned short* Ab = Vb + BSD;

    dim3 gg(Dsz / 128, Msz / 128);  // (16, 32)
    gemm_xt<0><<<gg, 256, 0, stream>>>(X, Wq, nullptr, Qb, Msz, Dsz, Dsz);
    gemm_xt<0><<<gg, 256, 0, stream>>>(X, Wk, kout,    Kb, Msz, Dsz, Dsz);
    gemm_xt<0><<<gg, 256, 0, stream>>>(X, Wv, vout,    Vb, Msz, Dsz, Dsz);

    attn_fwd<<<dim3(Ssz / 64, Bsz * Hsz), 128, 0, stream>>>(Qb, Kb, Vb, Ab);

    gemm_xt<1><<<gg, 256, 0, stream>>>(Ab, Wo, out, nullptr, Msz, Dsz, Dsz);
}
